// CISTransformerDecoder_63170378989582
// MI455X (gfx1250) — matrix operand order repaired and checked by reading, not compiled, not run on hardware
//
#include <hip/hip_runtime.h>
#include <math.h>

#define BSZ 2
#define NQ 1024
#define DM 256
#define NH 8
#define HD 32
#define NLV 4
#define NPT 4
#define NLAYERS 6
#define DFF 1024
#define LEN 21760   // 128*128 + 64*64 + 32*32 + 16*16

typedef __attribute__((ext_vector_type(16))) __bf16          v16bf;
typedef __attribute__((ext_vector_type(16))) unsigned short  v16u;
typedef __attribute__((ext_vector_type(8)))  float           v8f;

union BfVec { v16u u; v16bf b; };

__device__ __forceinline__ unsigned short f2bf(float f) {
  unsigned int u = __float_as_uint(f);
  u += 0x7FFFu + ((u >> 16) & 1u);            // round-to-nearest-even
  return (unsigned short)(u >> 16);
}
__device__ __forceinline__ unsigned int pack2bf(float lo, float hi) {
  return (unsigned int)f2bf(lo) | ((unsigned int)f2bf(hi) << 16);
}

// A-side WMMA fragment layout: AS[m/16][k/32][lane][elem], lane = half*16+m%16,
// elem i <-> k per the 16-bit A 16x32 table: k = (i<8?0:16) + half*8 + (i&7).
__device__ __forceinline__ size_t aswz_off(int m, int k, int Kd) {
  int mt = m >> 4, mm = m & 15;
  int kt = k >> 5, kk = k & 31;
  int half = (kk >> 3) & 1;
  int i = (kk & 7) | ((kk & 16) >> 1);
  return (((size_t)(mt * (Kd >> 5) + kt) * 32) + (size_t)(half * 16 + mm)) * 16 + i;
}

// ---------------------------------------------------------------------------
// Fragment-direct GEMM: C = act(A @ W + bias).  A,B pre-swizzled bf16 in
// global memory; per K-step each wave does 3x 32B coalesced fragment loads
// (global_load_b128 pairs) and 2 WMMAs.  No LDS, no barriers.
// Block = 256 thr = 8 waves (4M x 2N); block tile 64x64.  M%64,N%64,K%32 == 0.
// swout=0: fp32 row-major C.  swout=1: bf16 A-swizzled Csw (feeds next GEMM).
// ---------------------------------------------------------------------------
__global__ __launch_bounds__(256) void gemm_wmma_frag(
    const unsigned short* __restrict__ AS, const unsigned short* __restrict__ BS,
    const float* __restrict__ bias, float* __restrict__ C,
    unsigned short* __restrict__ Csw,
    int M, int N, int K, int act, int swout) {
  const int tid  = threadIdx.x;
  const int lane = tid & 31;
  const int wv   = tid >> 5;
  const int tr   = wv & 3;
  const int tc0  = (wv >> 2) * 2;
  const int hl   = lane >> 4;
  const int mn   = lane & 15;
  const int KT   = K >> 5;
  const int mt   = blockIdx.y * 4 + tr;
  const int nt0  = blockIdx.x * 4 + tc0;

  const v16u* Af = (const v16u*)AS + (size_t)mt * KT * 32 + lane;
  const v16u* B0 = (const v16u*)BS + (size_t)(nt0 + 0) * KT * 32 + lane;
  const v16u* B1 = (const v16u*)BS + (size_t)(nt0 + 1) * KT * 32 + lane;

  v8f acc0 = {};
  v8f acc1 = {};
  for (int kt = 0; kt < KT; ++kt) {
    BfVec a, b0, b1;
    a.u  = Af[(size_t)kt * 32];
    b0.u = B0[(size_t)kt * 32];
    b1.u = B1[(size_t)kt * 32];
    acc0 = __builtin_amdgcn_wmma_f32_16x16x32_bf16(
        false, a.b, false, b0.b, (short)0, acc0, false, false);
    acc1 = __builtin_amdgcn_wmma_f32_16x16x32_bf16(
        false, a.b, false, b1.b, (short)0, acc1, false, false);
  }

#pragma unroll
  for (int t = 0; t < 2; ++t) {
    const int col = blockIdx.x * 64 + (tc0 + t) * 16 + mn;
    const float bv = bias[col];
    const v8f acc = t ? acc1 : acc0;
#pragma unroll
    for (int r = 0; r < 8; ++r) {              // D: VGPR r -> M = r + 8*half
      int row = blockIdx.y * 64 + tr * 16 + hl * 8 + r;
      float v = acc[r] + bv;
      if (act) v = fmaxf(v, 0.0f);
      if (swout == 0) C[(size_t)row * N + col] = v;
      else            Csw[aswz_off(row, col, N)] = f2bf(v);
    }
  }
}

// ---------------------------------------------------------------------------
// Weight swizzle: W[K][N] fp32 -> BS[n/16][k/32][lane][elem] bf16,
// lane = half*16 + n%16, elem i <-> k = (i<8?0:16) + half*8 + (i&7).  Batched over blockIdx.y.
// ---------------------------------------------------------------------------
__global__ void swizzleB_kernel(const float* __restrict__ W,
                                unsigned short* __restrict__ BS, int K, int N) {
  int p = blockIdx.x * blockDim.x + threadIdx.x;
  if (p >= (K * N) >> 1) return;
  const float* Wm = W + (size_t)blockIdx.y * K * N;
  unsigned short* Bm = BS + (size_t)blockIdx.y * K * N;
  int n  = p % N;
  int k  = (p / N) * 2;
  float b0 = Wm[(size_t)k * N + n];
  float b1 = Wm[(size_t)(k + 1) * N + n];
  int nt = n >> 4, nn = n & 15;
  int kt = k >> 5, kk = k & 31;
  int half = (kk >> 3) & 1;
  int i = (kk & 7) | ((kk & 16) >> 1);          // even
  size_t off = (((size_t)(nt * (K >> 5) + kt) * 32) + (size_t)(half * 16 + nn)) * 16 + i;
  *reinterpret_cast<unsigned int*>(&Bm[off]) = pack2bf(b0, b1);
}

// ---------------------------------------------------------------------------
// Fused flatten(+pos+level_embed) -> bf16 A-swizzled value_in
// ---------------------------------------------------------------------------
__global__ void flatten_swz_kernel(const float* __restrict__ m0, const float* __restrict__ m1,
                                   const float* __restrict__ m2, const float* __restrict__ m3,
                                   const float* __restrict__ p0, const float* __restrict__ p1,
                                   const float* __restrict__ p2, const float* __restrict__ p3,
                                   const float* __restrict__ le,
                                   unsigned short* __restrict__ out_sw) {
  int p = blockIdx.x * blockDim.x + threadIdx.x;
  if (p >= BSZ * LEN * (DM / 2)) return;
  int c = (p & 127) * 2;
  int rest = p >> 7;
  int pos = rest % LEN;
  int b = rest / LEN;
  int l, hw, HW;
  if (pos < 16384)      { l = 0; hw = pos;         HW = 16384; }
  else if (pos < 20480) { l = 1; hw = pos - 16384; HW = 4096;  }
  else if (pos < 21504) { l = 2; hw = pos - 20480; HW = 1024;  }
  else                  { l = 3; hw = pos - 21504; HW = 256;   }
  const float* mm = (l == 0) ? m0 : (l == 1) ? m1 : (l == 2) ? m2 : m3;
  const float* pp = (l == 0) ? p0 : (l == 1) ? p1 : (l == 2) ? p2 : p3;
  size_t s0 = ((size_t)b * DM + c) * HW + hw;
  size_t s1 = s0 + HW;
  float v0 = mm[s0] + pp[s0] + le[l * DM + c];
  float v1 = mm[s1] + pp[s1] + le[l * DM + c + 1];
  size_t off = aswz_off(b * LEN + pos, c, DM);
  *reinterpret_cast<unsigned int*>(&out_sw[off]) = pack2bf(v0, v1);
}

// sine positional embedding of sigmoid(ref) -> bf16 A-swizzled qs (K=512)
__global__ void sine_swz_kernel(const float* __restrict__ ref,
                                unsigned short* __restrict__ qs_sw) {
  int p = blockIdx.x * blockDim.x + threadIdx.x;
  if (p >= BSZ * NQ * 256) return;
  int j = (p & 255) * 2;                 // even: (sin, cos) pair, same freq
  int bq = p >> 8;
  int comp = j >> 7;
  int wi = j & 127;                      // even
  const int cmap[4] = {1, 0, 2, 3};      // y, x, w, h
  float v = ref[bq * 4 + cmap[comp]];
  v = 1.0f / (1.0f + expf(-v));
  float ang = v * 6.283185307179586f / powf(10000.0f, (float)wi / 128.0f);
  size_t off = aswz_off(bq, j, 512);
  *reinterpret_cast<unsigned int*>(&qs_sw[off]) = pack2bf(sinf(ang), cosf(ang));
}

// q = src + query_pos, emitted bf16 A-swizzled (K=256)
__global__ void add_swz_kernel(const float* __restrict__ a, const float* __restrict__ b,
                               unsigned short* __restrict__ out_sw) {
  int p = blockIdx.x * blockDim.x + threadIdx.x;
  if (p >= BSZ * NQ * (DM / 2)) return;
  int k = (p & 127) * 2;
  int m = p >> 7;
  const float2 av = *(const float2*)(a + (size_t)m * DM + k);
  const float2 bv = *(const float2*)(b + (size_t)m * DM + k);
  size_t off = aswz_off(m, k, DM);
  *reinterpret_cast<unsigned int*>(&out_sw[off]) = pack2bf(av.x + bv.x, av.y + bv.y);
}

// valid ratios from padding masks: vr[b][l] = (vw, vh)
__global__ void vr_kernel(const unsigned char* __restrict__ k0, const unsigned char* __restrict__ k1,
                          const unsigned char* __restrict__ k2, const unsigned char* __restrict__ k3,
                          float* __restrict__ vr) {
  int t = threadIdx.x;
  if (t >= BSZ * NLV) return;
  int b = t >> 2, l = t & 3;
  const int Hs[4] = {128, 64, 32, 16};
  const unsigned char* mk = (l == 0) ? k0 : (l == 1) ? k1 : (l == 2) ? k2 : k3;
  int H = Hs[l], W = Hs[l];
  size_t base = (size_t)b * H * W;
  int ch = 0, cw = 0;
  for (int i = 0; i < H; ++i) ch += (mk[base + (size_t)i * W] == 0);
  for (int i = 0; i < W; ++i) cw += (mk[base + i] == 0);
  vr[(b * NLV + l) * 2 + 0] = (float)cw / (float)W;
  vr[(b * NLV + l) * 2 + 1] = (float)ch / (float)H;
}

__global__ void copy_kernel(const float* __restrict__ a, float* __restrict__ o, int n) {
  int i = blockIdx.x * blockDim.x + threadIdx.x;
  if (i < n) o[i] = a[i];
}

// softmax over NL*NP=16 contiguous logits, in place
__global__ void softmax16_kernel(float* __restrict__ aw, int rows) {
  int t = blockIdx.x * blockDim.x + threadIdx.x;
  if (t >= rows) return;
  float* p = aw + (size_t)t * 16;
  float mx = p[0];
#pragma unroll
  for (int i = 1; i < 16; ++i) mx = fmaxf(mx, p[i]);
  float s = 0.f;
  float e[16];
#pragma unroll
  for (int i = 0; i < 16; ++i) { e[i] = expf(p[i] - mx); s += e[i]; }
  float inv = 1.0f / s;
#pragma unroll
  for (int i = 0; i < 16; ++i) p[i] = e[i] * inv;
}

// ---------------------------------------------------------------------------
// ms_deform_core: one wave per (b,q,head); lane = channel (HD=32 = wave32).
// Output written directly as bf16 A-swizzled (feeds the op-projection GEMM).
// ---------------------------------------------------------------------------
__global__ __launch_bounds__(256) void msdeform_kernel(
    const float* __restrict__ value, const float* __restrict__ so,
    const float* __restrict__ aw, const float* __restrict__ ref,
    const float* __restrict__ vr, unsigned short* __restrict__ out_sw) {
  const int Hs[4] = {128, 64, 32, 16};
  const int St[4] = {0, 16384, 20480, 21504};
  int gid  = blockIdx.x * 8 + (threadIdx.x >> 5);
  int lane = threadIdx.x & 31;
  int b = gid / (NQ * NH);
  int rem = gid - b * (NQ * NH);
  int q = rem >> 3;
  int h = rem & 7;
  const float* rp = ref + (size_t)(b * NQ + q) * 4;
  float acc = 0.f;
#pragma unroll
  for (int l = 0; l < NLV; ++l) {
    int Hh = Hs[l], Ww = Hs[l], st = St[l];
    float vw = vr[(b * NLV + l) * 2 + 0];
    float vh = vr[(b * NLV + l) * 2 + 1];
    float rx = rp[0] * vw, ry = rp[1] * vh;
    float rw = rp[2] * vw, rh = rp[3] * vh;
#pragma unroll
    for (int p = 0; p < NPT; ++p) {
      size_t sbase = ((((size_t)(b * NQ + q) * NH + h) * NLV + l) * NPT + p);
      float sox = so[sbase * 2], soy = so[sbase * 2 + 1];
      float locx = rx + sox * (1.0f / NPT) * rw * 0.5f;
      float locy = ry + soy * (1.0f / NPT) * rh * 0.5f;
      float x = locx * (float)Ww - 0.5f;
      float y = locy * (float)Hh - 0.5f;
      float x0 = floorf(x), y0 = floorf(y);
      float lx = x - x0, ly = y - y0;
      int ix = (int)x0, iy = (int)y0;
      float a = aw[sbase];
      float w00 = (1.f - lx) * (1.f - ly) * a;
      float w10 = lx * (1.f - ly) * a;
      float w01 = (1.f - lx) * ly * a;
      float w11 = lx * ly * a;
#define MSD_CORNER(XI, YI, WGT)                                                   \
      if ((XI) >= 0 && (XI) < Ww && (YI) >= 0 && (YI) < Hh) {                     \
        size_t vi = (((size_t)b * LEN + st + (size_t)(YI) * Ww + (XI)) * NH + h)  \
                    * HD + lane;                                                  \
        acc += (WGT) * value[vi];                                                 \
      }
      MSD_CORNER(ix,     iy,     w00)
      MSD_CORNER(ix + 1, iy,     w10)
      MSD_CORNER(ix,     iy + 1, w01)
      MSD_CORNER(ix + 1, iy + 1, w11)
#undef MSD_CORNER
    }
  }
  out_sw[aswz_off(b * NQ + q, h * HD + lane, DM)] = f2bf(acc);
}

// fused residual + layernorm: src = LN(src + add) * s + b (row = 256), plus a
// bf16 A-swizzled copy of the result for the following GEMM.
__global__ __launch_bounds__(256) void ln_kernel(float* __restrict__ src,
                                                 const float* __restrict__ add,
                                                 const float* __restrict__ s,
                                                 const float* __restrict__ bb,
                                                 unsigned short* __restrict__ sw) {
  __shared__ float red[256];
  int row = blockIdx.x, t = threadIdx.x;
  size_t o = (size_t)row * DM + t;
  float x = src[o] + add[o];
  red[t] = x;
  __syncthreads();
  for (int off = 128; off > 0; off >>= 1) {
    if (t < off) red[t] += red[t + off];
    __syncthreads();
  }
  float mean = red[0] * (1.0f / DM);
  __syncthreads();
  float d = x - mean;
  red[t] = d * d;
  __syncthreads();
  for (int off = 128; off > 0; off >>= 1) {
    if (t < off) red[t] += red[t + off];
    __syncthreads();
  }
  float var = red[0] * (1.0f / DM);
  float v = d * rsqrtf(var + 1e-5f) * s[t] + bb[t];
  src[o] = v;
  sw[aswz_off(row, t, DM)] = f2bf(v);
}

// ---------------------------------------------------------------------------
extern "C" void kernel_launch(void* const* d_in, const int* in_sizes, int n_in,
                              void* d_out, int out_size, void* d_ws, size_t ws_size,
                              hipStream_t stream) {
  (void)in_sizes; (void)n_in; (void)out_size; (void)ws_size;
  const float* tgts = (const float*)d_in[0];
  const float* ref  = (const float*)d_in[2];
  const unsigned char* mk0 = (const unsigned char*)d_in[3];
  const float* mem0 = (const float*)d_in[4];
  const float* pm0  = (const float*)d_in[5];
  const unsigned char* mk1 = (const unsigned char*)d_in[6];
  const float* mem1 = (const float*)d_in[7];
  const float* pm1  = (const float*)d_in[8];
  const unsigned char* mk2 = (const unsigned char*)d_in[9];
  const float* mem2 = (const float*)d_in[10];
  const float* pm2  = (const float*)d_in[11];
  const unsigned char* mk3 = (const unsigned char*)d_in[12];
  const float* mem3 = (const float*)d_in[13];
  const float* pm3  = (const float*)d_in[14];
  const float* le    = (const float*)d_in[15];
  const float* rp_w1 = (const float*)d_in[16];
  const float* rp_b1 = (const float*)d_in[17];
  const float* rp_w2 = (const float*)d_in[18];
  const float* rp_b2 = (const float*)d_in[19];
  const float* so_w  = (const float*)d_in[20];
  const float* so_b  = (const float*)d_in[21];
  const float* aw_w  = (const float*)d_in[22];
  const float* aw_b  = (const float*)d_in[23];
  const float* vp_w  = (const float*)d_in[24];
  const float* vp_b  = (const float*)d_in[25];
  const float* op_w  = (const float*)d_in[26];
  const float* op_b  = (const float*)d_in[27];
  const float* n1_s  = (const float*)d_in[28];
  const float* n1_b  = (const float*)d_in[29];
  const float* f1_w  = (const float*)d_in[30];
  const float* f1_b  = (const float*)d_in[31];
  const float* f2_w  = (const float*)d_in[32];
  const float* f2_b  = (const float*)d_in[33];
  const float* n2_s  = (const float*)d_in[34];
  const float* n2_b  = (const float*)d_in[35];

  float* src = (float*)d_out;   // running residual stream, (2,1024,256)

  // ---- workspace carve (32B-aligned regions; fp32 then bf16/ushort) ----
  char* base = (char*)d_ws;
  size_t off = 0;
  auto allocF = [&](size_t n) -> float* {
    float* p = (float*)(base + off); off += n * 4; off = (off + 31) & ~(size_t)31; return p;
  };
  auto allocU = [&](size_t n) -> unsigned short* {
    unsigned short* p = (unsigned short*)(base + off); off += n * 2; off = (off + 31) & ~(size_t)31; return p;
  };
  float* value  = allocF((size_t)BSZ * LEN * DM);   // fp32, read by msdeform
  float* qpos   = allocF((size_t)BSZ * NQ * DM);
  float* sobuf  = allocF((size_t)BSZ * NQ * DM);
  float* awbuf  = allocF((size_t)BSZ * NQ * 128);
  float* proj   = allocF((size_t)BSZ * NQ * DM);
  float* vrp    = allocF(16);
  unsigned short* value_sw = allocU((size_t)BSZ * LEN * DM);
  unsigned short* qs_sw    = allocU((size_t)BSZ * NQ * 512);
  unsigned short* qhid_sw  = allocU((size_t)BSZ * NQ * DM);
  unsigned short* qbuf_sw  = allocU((size_t)BSZ * NQ * DM);
  unsigned short* attn_sw  = allocU((size_t)BSZ * NQ * DM);
  unsigned short* srcn_sw  = allocU((size_t)BSZ * NQ * DM);
  unsigned short* ffh_sw   = allocU((size_t)BSZ * NQ * DFF);
  unsigned short* rp1_sw   = allocU((size_t)512 * DM);
  unsigned short* rp2_sw   = allocU((size_t)DM * DM);
  unsigned short* so_sw    = allocU((size_t)NLAYERS * DM * 256);
  unsigned short* aw_sw    = allocU((size_t)NLAYERS * DM * 128);
  unsigned short* vp_sw    = allocU((size_t)NLAYERS * DM * DM);
  unsigned short* op_sw    = allocU((size_t)NLAYERS * DM * DM);
  unsigned short* f1_sw    = allocU((size_t)NLAYERS * DM * DFF);
  unsigned short* f2_sw    = allocU((size_t)NLAYERS * DFF * DM);

  const int NTOK = BSZ * NQ;            // 2048 query rows
  const int NELEM = NTOK * DM;          // 524288
  const int MV = BSZ * LEN;             // 43520 value rows

  // ---- one-time staging ----
  copy_kernel<<<(NELEM + 255) / 256, 256, 0, stream>>>(tgts, src, NELEM);
  flatten_swz_kernel<<<(MV * (DM / 2) + 255) / 256, 256, 0, stream>>>(
      mem0, mem1, mem2, mem3, pm0, pm1, pm2, pm3, le, value_sw);
  vr_kernel<<<1, 32, 0, stream>>>(mk0, mk1, mk2, mk3, vrp);
  // weights -> swizzled bf16 (once per call)
  swizzleB_kernel<<<dim3((512 * DM / 2 + 255) / 256, 1), 256, 0, stream>>>(rp_w1, rp1_sw, 512, DM);
  swizzleB_kernel<<<dim3((DM * DM / 2 + 255) / 256, 1), 256, 0, stream>>>(rp_w2, rp2_sw, DM, DM);
  swizzleB_kernel<<<dim3((DM * 256 / 2 + 255) / 256, NLAYERS), 256, 0, stream>>>(so_w, so_sw, DM, 256);
  swizzleB_kernel<<<dim3((DM * 128 / 2 + 255) / 256, NLAYERS), 256, 0, stream>>>(aw_w, aw_sw, DM, 128);
  swizzleB_kernel<<<dim3((DM * DM / 2 + 255) / 256, NLAYERS), 256, 0, stream>>>(vp_w, vp_sw, DM, DM);
  swizzleB_kernel<<<dim3((DM * DM / 2 + 255) / 256, NLAYERS), 256, 0, stream>>>(op_w, op_sw, DM, DM);
  swizzleB_kernel<<<dim3((DM * DFF / 2 + 255) / 256, NLAYERS), 256, 0, stream>>>(f1_w, f1_sw, DM, DFF);
  swizzleB_kernel<<<dim3((DFF * DM / 2 + 255) / 256, NLAYERS), 256, 0, stream>>>(f2_w, f2_sw, DFF, DM);
  // query_pos = relu(sine(sigmoid(ref)) @ rp_w1 + b1) @ rp_w2 + b2
  sine_swz_kernel<<<(BSZ * NQ * 256 + 255) / 256, 256, 0, stream>>>(ref, qs_sw);
  gemm_wmma_frag<<<dim3(DM / 64, NTOK / 64), 256, 0, stream>>>(
      qs_sw, rp1_sw, rp_b1, nullptr, qhid_sw, NTOK, DM, 512, 1, 1);
  gemm_wmma_frag<<<dim3(DM / 64, NTOK / 64), 256, 0, stream>>>(
      qhid_sw, rp2_sw, rp_b2, qpos, nullptr, NTOK, DM, DM, 0, 0);

  for (int i = 0; i < NLAYERS; ++i) {
    // value = value_in @ vp_w[i] + vp_b[i]   (43520 x 256 x 256)
    gemm_wmma_frag<<<dim3(DM / 64, MV / 64), 256, 0, stream>>>(
        value_sw, vp_sw + (size_t)i * DM * DM, vp_b + (size_t)i * DM,
        value, nullptr, MV, DM, DM, 0, 0);
    // q = src + query_pos  (swizzled for the two query-side GEMMs)
    add_swz_kernel<<<(NTOK * (DM / 2) + 255) / 256, 256, 0, stream>>>(src, qpos, qbuf_sw);
    // sampling offsets + attention logits
    gemm_wmma_frag<<<dim3(256 / 64, NTOK / 64), 256, 0, stream>>>(
        qbuf_sw, so_sw + (size_t)i * DM * 256, so_b + (size_t)i * 256,
        sobuf, nullptr, NTOK, 256, DM, 0, 0);
    gemm_wmma_frag<<<dim3(128 / 64, NTOK / 64), 256, 0, stream>>>(
        qbuf_sw, aw_sw + (size_t)i * DM * 128, aw_b + (size_t)i * 128,
        awbuf, nullptr, NTOK, 128, DM, 0, 0);
    softmax16_kernel<<<(NTOK * NH + 255) / 256, 256, 0, stream>>>(awbuf, NTOK * NH);
    // deformable sampling -> swizzled attn
    msdeform_kernel<<<(NTOK * NH) / 8, 256, 0, stream>>>(
        value, sobuf, awbuf, ref, vrp, attn_sw);
    // output projection + residual LN (LN emits swizzled src for FFN)
    gemm_wmma_frag<<<dim3(DM / 64, NTOK / 64), 256, 0, stream>>>(
        attn_sw, op_sw + (size_t)i * DM * DM, op_b + (size_t)i * DM,
        proj, nullptr, NTOK, DM, DM, 0, 0);
    ln_kernel<<<NTOK, 256, 0, stream>>>(src, proj, n1_s + (size_t)i * DM,
                                        n1_b + (size_t)i * DM, srcn_sw);
    // FFN
    gemm_wmma_frag<<<dim3(DFF / 64, NTOK / 64), 256, 0, stream>>>(
        srcn_sw, f1_sw + (size_t)i * DM * DFF, f1_b + (size_t)i * DFF,
        nullptr, ffh_sw, NTOK, DFF, DM, 1, 1);
    gemm_wmma_frag<<<dim3(DM / 64, NTOK / 64), 256, 0, stream>>>(
        ffh_sw, f2_sw + (size_t)i * DFF * DM, f2_b + (size_t)i * DM,
        proj, nullptr, NTOK, DM, DFF, 0, 0);
    ln_kernel<<<NTOK, 256, 0, stream>>>(src, proj, n2_s + (size_t)i * DM,
                                        n2_b + (size_t)i * DM, srcn_sw);
  }
}
